// MoELayer_71674414235937
// MI455X (gfx1250) — matrix operand order repaired and checked by reading, not compiled, not run on hardware
//
#include <hip/hip_runtime.h>
#include <hip/hip_bf16.h>

typedef unsigned short u16;
typedef __attribute__((ext_vector_type(16))) __bf16 v16bf;
typedef __attribute__((ext_vector_type(8)))  float  v8f;

#define BB 4
#define SS 2048
#define DD 1024
#define II 4096
#define EE 8
#define TT (BB*SS)          // 8192 tokens
#define XP 1032             // x-tile LDS row stride (bf16 elems), 2064B = 129*16
#define HP 136              // h-tile LDS row stride (bf16 elems), 272B  = 17*16
#define MT 32               // tokens per block

__device__ __forceinline__ u16 f2bf(float f) {
  union { float f; unsigned u; } c; c.f = f;
  unsigned r = c.u + 0x7FFFu + ((c.u >> 16) & 1u);   // round-to-nearest-even
  return (u16)(r >> 16);
}

union FragBF { uint4 u4[2]; v16bf v; };

// ---------------------------------------------------------------- zero output
__global__ void zero_f32_kernel(float* __restrict__ p, int n) {
  int i = blockIdx.x * blockDim.x + threadIdx.x;
  if (i < n) p[i] = 0.0f;
}

// ---------------------------------------------------------------- x -> bf16
__global__ void cvt_x_kernel(const float* __restrict__ x, u16* __restrict__ xb, int n) {
  int i = blockIdx.x * blockDim.x + threadIdx.x;
  if (i < n) xb[i] = f2bf(x[i]);
}

// ------------------------------------------- W[e] (R x C) -> Wt[e] (C x R) bf16
__global__ void cvt_transpose_kernel(const float* __restrict__ in, u16* __restrict__ out,
                                     int R, int C) {
  __shared__ float tile[32][33];
  const int e  = blockIdx.z;
  const float* src = in  + (size_t)e * R * C;
  u16*         dst = out + (size_t)e * R * C;
  const int c0 = blockIdx.x * 32, r0 = blockIdx.y * 32;
  for (int j = threadIdx.y; j < 32; j += 8)
    tile[j][threadIdx.x] = src[(size_t)(r0 + j) * C + c0 + threadIdx.x];
  __syncthreads();
  for (int j = threadIdx.y; j < 32; j += 8)
    dst[(size_t)(c0 + j) * R + r0 + threadIdx.x] = f2bf(tile[threadIdx.x][j]);
}

// ---------------------------------------------------------------- router
// one wave (32 lanes) per token; 8 waves / block
__global__ __launch_bounds__(256)
void router_kernel(const float* __restrict__ x, const float* __restrict__ Wr,
                   const float* __restrict__ br, int* __restrict__ route_e,
                   float* __restrict__ route_g) {
  const int lane = threadIdx.x & 31;
  const int wv   = threadIdx.x >> 5;
  const int t    = blockIdx.x * 8 + wv;
  float acc[EE];
#pragma unroll
  for (int e = 0; e < EE; ++e) acc[e] = 0.0f;
  const float* xp = x + (size_t)t * DD;
  for (int j = lane; j < DD; j += 32) {
    float xv = xp[j];
#pragma unroll
    for (int e = 0; e < EE; ++e) acc[e] += xv * Wr[e * DD + j];
  }
#pragma unroll
  for (int e = 0; e < EE; ++e)
#pragma unroll
    for (int off = 16; off > 0; off >>= 1) acc[e] += __shfl_xor(acc[e], off, 32);
  if (lane == 0) {
    float lg[EE];
#pragma unroll
    for (int e = 0; e < EE; ++e) lg[e] = acc[e] + br[e];
    float v1 = -1e30f; int e1 = 0;
#pragma unroll
    for (int e = 0; e < EE; ++e) if (lg[e] > v1) { v1 = lg[e]; e1 = e; }
    float v2 = -1e30f; int e2 = (e1 == 0) ? 1 : 0;
#pragma unroll
    for (int e = 0; e < EE; ++e) if (e != e1 && lg[e] > v2) { v2 = lg[e]; e2 = e; }
    float g1 = 1.0f / (1.0f + expf(v2 - v1));      // softmax over {v1,v2}
    route_e[t] = e1 | (e2 << 8);
    route_g[2 * t]     = g1;
    route_g[2 * t + 1] = 1.0f - g1;
  }
}

// --------------------------------------- deterministic per-expert list build
__global__ __launch_bounds__(256)
void build_lists_kernel(const int* __restrict__ route_e, const float* __restrict__ route_g,
                        int* __restrict__ ltok, float* __restrict__ lgate,
                        int* __restrict__ counts) {
  const int e    = blockIdx.x;
  const int tid  = threadIdx.x;
  const int lane = tid & 31;
  const int wv   = tid >> 5;
  __shared__ int wsum[8];
  __shared__ int running_s;
  if (tid == 0) running_s = 0;
  __syncthreads();
  int* lt = ltok + (size_t)e * TT;
  float* lgp = lgate + (size_t)e * TT;
  for (int base = 0; base < TT; base += 256) {
    const int t = base + tid;
    const int pk = route_e[t];
    const int e1 = pk & 0xFF, e2 = (pk >> 8) & 0xFF;
    const bool s1 = (e1 == e), s2 = (e2 == e);
    const bool sel = s1 || s2;
    const float g = s1 ? route_g[2 * t] : route_g[2 * t + 1];
    unsigned m = __builtin_amdgcn_ballot_w32(sel);
    int my = __popc(m & ((1u << lane) - 1u));
    if (lane == 0) wsum[wv] = __popc(m);
    __syncthreads();
    int off = running_s;
    for (int w = 0; w < wv; ++w) off += wsum[w];
    if (sel) { lt[off + my] = t; lgp[off + my] = g; }
    __syncthreads();
    if (tid == 0) {
      int tot = 0;
#pragma unroll
      for (int w = 0; w < 8; ++w) tot += wsum[w];
      running_s += tot;
    }
    __syncthreads();
  }
  if (tid == 0) counts[e] = running_s;
}

// ------------------------------------------------- fused top-k expert FFN
// grid: (TT/32, EE), block 512 (16 waves). Each block: 32 tokens of expert e.
// Wave w: GEMM1 computes h row-tile (w>>3), col-tile (w&7) of the 32x128 chunk;
//         GEMM2 owns 64 D-columns (nbase = w*64), both 16-row tiles.
__global__ __launch_bounds__(512)
void moe_ffn_kernel(const u16* __restrict__ xbf, const u16* __restrict__ w1t,
                    const u16* __restrict__ w2t, const float* __restrict__ b1,
                    const float* __restrict__ b2, const int* __restrict__ ltok,
                    const float* __restrict__ lgate, const int* __restrict__ counts,
                    float* __restrict__ out) {
  const int e   = blockIdx.y;
  const int blk = blockIdx.x;
  const int cnt = counts[e];
  if (blk * MT >= cnt) return;

  __shared__ __align__(16) u16 xs[MT * XP];
  __shared__ __align__(16) u16 hs[MT * HP];
  __shared__ int   tok_s[MT];
  __shared__ float gate_s[MT];

  const int tid  = threadIdx.x;
  const int lane = tid & 31;
  const int wv   = tid >> 5;          // 0..15
  const int half = lane >> 4;
  const int ln   = lane & 15;

  if (tid < MT) {
    const int idx = blk * MT + tid;
    if (idx < cnt) {
      tok_s[tid]  = ltok[(size_t)e * TT + idx];
      gate_s[tid] = lgate[(size_t)e * TT + idx];
    } else { tok_s[tid] = -1; gate_s[tid] = 0.0f; }
  }
  __syncthreads();

  // stage 32 x 1024 bf16 x-rows into LDS: async global->LDS (ASYNCcnt path),
  // zero-fill padding rows with plain LDS stores.
#pragma unroll
  for (int it = 0; it < 8; ++it) {
    const int idx = it * 512 + tid;
    const int r   = idx >> 7;         // 0..31
    const int c8  = idx & 127;        // 0..127
    const int tkn = tok_s[r];
    if (tkn >= 0) {
      const unsigned voff = (unsigned)(((unsigned)tkn * DD + c8 * 8) * 2);
      const unsigned ldsa = (unsigned)(size_t)(&xs[r * XP + c8 * 8]);
      asm volatile("global_load_async_to_lds_b128 %0, %1, %2 offset:0"
                   :: "v"(ldsa), "v"(voff), "s"(xbf) : "memory");
    } else {
      *(uint4*)(&xs[r * XP + c8 * 8]) = make_uint4(0u, 0u, 0u, 0u);
    }
  }
  asm volatile("s_wait_asynccnt 0x0" ::: "memory");
  __syncthreads();

  v8f acc[8];                         // [row-tile 0: t=0..3][row-tile 1: t=4..7]
#pragma unroll
  for (int t = 0; t < 8; ++t)
#pragma unroll
    for (int v = 0; v < 8; ++v) acc[t][v] = 0.0f;

  const int rt1 = wv >> 3;            // GEMM1 row-tile (0/1)
  const int ct1 = wv & 7;             // GEMM1 col-tile (0..7)

  for (int ic = 0; ic < II / 128; ++ic) {
    const int icbase = ic * 128;
    // prefetch next chunk of this wave's W2t rows into cache
    if (ic + 1 < II / 128)
      __builtin_prefetch(w2t + ((size_t)e * DD + wv * 64 + ln) * II + icbase + 128, 0, 0);

    // ---- GEMM1: one 16x16 h tile per wave, K over D=1024
    v8f hacc = {0.f, 0.f, 0.f, 0.f, 0.f, 0.f, 0.f, 0.f};
    const int icol = icbase + ct1 * 16 + ln;              // B-matrix column (N)
    const u16* bp1 = w1t + ((size_t)e * II + icol) * DD + 8 * half;
    const int arow = (rt1 * 16 + ln) * XP;
    for (int kk = 0; kk < DD; kk += 32) {
      FragBF a, b;
      a.u4[0] = *(const uint4*)(&xs[arow + kk + 8 * half]);
      a.u4[1] = *(const uint4*)(&xs[arow + kk + 16 + 8 * half]);
      b.u4[0] = *(const uint4*)(bp1 + kk);
      b.u4[1] = *(const uint4*)(bp1 + kk + 16);
      hacc = __builtin_amdgcn_wmma_f32_16x16x32_bf16(false, a.v, false, b.v,
                                                     (short)0, hacc, false, false);
    }
    // bias + exact GELU, write bf16 h tile to LDS
    const float bias = b1[e * II + icol];
    const int hcol = ct1 * 16 + ln;
#pragma unroll
    for (int v = 0; v < 8; ++v) {
      float hv = hacc[v] + bias;
      hv = 0.5f * hv * (1.0f + erff(hv * 0.70710678118654752f));
      hs[(rt1 * 16 + v + 8 * half) * HP + hcol] = f2bf(hv);
    }
    __syncthreads();

    // ---- GEMM2: acc += h(32x128) @ W2t-tiles; B fragment shared by 2 row-tiles
#pragma unroll
    for (int kk = 0; kk < 128; kk += 32) {
      FragBF a0, a1;
      a0.u4[0] = *(const uint4*)(&hs[ln * HP + kk + 8 * half]);
      a0.u4[1] = *(const uint4*)(&hs[ln * HP + kk + 16 + 8 * half]);
      a1.u4[0] = *(const uint4*)(&hs[(16 + ln) * HP + kk + 8 * half]);
      a1.u4[1] = *(const uint4*)(&hs[(16 + ln) * HP + kk + 16 + 8 * half]);
#pragma unroll
      for (int t = 0; t < 4; ++t) {
        const int dcol = wv * 64 + t * 16 + ln;
        const u16* bp2 = w2t + ((size_t)e * DD + dcol) * II + icbase + 8 * half;
        FragBF b;
        b.u4[0] = *(const uint4*)(bp2 + kk);
        b.u4[1] = *(const uint4*)(bp2 + kk + 16);
        acc[t]     = __builtin_amdgcn_wmma_f32_16x16x32_bf16(false, a0.v, false, b.v,
                                                             (short)0, acc[t], false, false);
        acc[4 + t] = __builtin_amdgcn_wmma_f32_16x16x32_bf16(false, a1.v, false, b.v,
                                                             (short)0, acc[4 + t], false, false);
      }
    }
    __syncthreads();
  }

  // ---- epilogue: +b2, scale by gate, scatter-accumulate into out
#pragma unroll
  for (int t = 0; t < 4; ++t) {
    const int dcol = wv * 64 + t * 16 + ln;
    const float bb = b2[e * DD + dcol];
#pragma unroll
    for (int r = 0; r < 2; ++r) {
#pragma unroll
      for (int v = 0; v < 8; ++v) {
        const int m   = r * 16 + v + 8 * half;
        const int tkn = tok_s[m];
        if (tkn >= 0)
          atomicAdd(&out[(size_t)tkn * DD + dcol], gate_s[m] * (acc[r * 4 + t][v] + bb));
      }
    }
  }
}

// ---------------------------------------------------------------- launcher
extern "C" void kernel_launch(void* const* d_in, const int* in_sizes, int n_in,
                              void* d_out, int out_size, void* d_ws, size_t ws_size,
                              hipStream_t stream) {
  const float* x  = (const float*)d_in[0];
  const float* Wr = (const float*)d_in[1];
  const float* br = (const float*)d_in[2];
  const float* W1 = (const float*)d_in[3];
  const float* b1 = (const float*)d_in[4];
  const float* W2 = (const float*)d_in[5];
  const float* b2 = (const float*)d_in[6];
  float* out = (float*)d_out;

  char* ws = (char*)d_ws;
  u16* xbf = (u16*)ws;            ws += (size_t)TT * DD * 2;
  u16* w1t = (u16*)ws;            ws += (size_t)EE * II * DD * 2;
  u16* w2t = (u16*)ws;            ws += (size_t)EE * DD * II * 2;
  int*   route_e = (int*)ws;      ws += (size_t)TT * 4;
  float* route_g = (float*)ws;    ws += (size_t)TT * 2 * 4;
  int*   ltok    = (int*)ws;      ws += (size_t)EE * TT * 4;
  float* lgate   = (float*)ws;    ws += (size_t)EE * TT * 4;
  int*   counts  = (int*)ws;      ws += (size_t)EE * 4;

  const int nOut = TT * DD;
  zero_f32_kernel<<<(nOut + 255) / 256, 256, 0, stream>>>(out, nOut);

  const int nX = TT * DD;
  cvt_x_kernel<<<(nX + 255) / 256, 256, 0, stream>>>(x, xbf, nX);

  // W1 [E,D,I] -> W1t [E,I,D];  W2 [E,I,D] -> W2t [E,D,I]
  cvt_transpose_kernel<<<dim3(II / 32, DD / 32, EE), dim3(32, 8), 0, stream>>>(W1, w1t, DD, II);
  cvt_transpose_kernel<<<dim3(DD / 32, II / 32, EE), dim3(32, 8), 0, stream>>>(W2, w2t, II, DD);

  router_kernel<<<TT / 8, 256, 0, stream>>>(x, Wr, br, route_e, route_g);
  build_lists_kernel<<<EE, 256, 0, stream>>>(route_e, route_g, ltok, lgate, counts);

  moe_ffn_kernel<<<dim3(TT / MT, EE), 512, 0, stream>>>(xbf, w1t, w2t, b1, b2,
                                                        ltok, lgate, counts, out);
}
